// GaussianSelfAttention_3676492005885
// MI455X (gfx1250) — matrix operand order repaired and checked by reading, not compiled, not run on hardware
//
#include <hip/hip_runtime.h>
#include <hip/hip_bf16.h>

typedef __attribute__((ext_vector_type(16))) __bf16 v16bf;
typedef __attribute__((ext_vector_type(8)))  float  v8f;

constexpr int kB  = 128;
constexpr int kP  = 196;
constexpr int kD  = 768;
constexpr int kGD = 14;
constexpr int kM  = kB * kP;              // 25088 rows
constexpr int kWELE = kD * kD;            // 589824 per weight matrix
constexpr size_t kSZ = (size_t)kM * kD;   // elements per q/k/v matrix
constexpr size_t kQKV_BYTE_OFF = (size_t)3 * kWELE * 2;  // bf16 weights region
constexpr int kXS_STRIDE = 776;           // padded LDS stride (bf16 units): 1552B = 16B-aligned rows, 4-bank skew

__device__ __forceinline__ unsigned int f2bf(float f) {
  unsigned int u = __float_as_uint(f);
  return (u + 0x7FFFu + ((u >> 16) & 1u)) >> 16;   // RNE fp32 -> bf16
}

// ---------------- Kernel 0: convert Wq/Wk/Wv fp32 -> bf16 (row-major) ----------------
__global__ __launch_bounds__(256) void k_convw(const float* __restrict__ Wq,
                                               const float* __restrict__ Wk,
                                               const float* __restrict__ Wv,
                                               unsigned short* __restrict__ out) {
  unsigned int idx = blockIdx.x * 256u + threadIdx.x;   // 0 .. 3*kWELE-1 exactly
  const float* src;
  unsigned int off;
  if (idx < (unsigned)kWELE)          { src = Wq; off = idx; }
  else if (idx < 2u * kWELE)          { src = Wk; off = idx - kWELE; }
  else                                { src = Wv; off = idx - 2u * kWELE; }
  out[idx] = (unsigned short)f2bf(src[off]);
}

// ---------------- Kernel 1: fused QKV GEMM via bf16 WMMA ----------------
// grid = (1568 M-tiles, 3 matrices), block = 256 threads = 8 waves.
// Each wave owns 6 adjacent 16-wide N-tiles (8*6*16 = 768 = full N).
__global__ __launch_bounds__(256) void k_gemm_qkv(const float* __restrict__ x,
                                                  const unsigned short* __restrict__ Wbf,
                                                  const float* __restrict__ bq,
                                                  const float* __restrict__ bk,
                                                  const float* __restrict__ bv,
                                                  float* __restrict__ qkv) {
  __shared__ __align__(16) unsigned short Xs[16 * kXS_STRIDE];
  const int mtile = blockIdx.x;
  const int mat   = blockIdx.y;
  const int t     = threadIdx.x;

  // Stage x[mtile*16 .. +16, 0:768] into LDS as bf16 (coalesced fp32 pair loads).
  const float* xrow = x + (size_t)mtile * 16 * kD;
  #pragma unroll
  for (int it = 0; it < 24; ++it) {
    int pi  = t + it * 256;              // pair index 0..6143
    int row = pi / 384;                  // 384 pairs per row
    int kp  = (pi - row * 384) * 2;
    float2 f2 = *(const float2*)(xrow + row * kD + kp);
    unsigned int pk = f2bf(f2.x) | (f2bf(f2.y) << 16);
    *(unsigned int*)(&Xs[row * kXS_STRIDE + kp]) = pk;
  }
  __syncthreads();

  const int lane = t & 31;
  const int wave = t >> 5;
  const int nl   = lane & 15;            // A row within tile == B column within tile
  const int hi   = lane >> 4;            // lane half selector

  const unsigned short* Wm = Wbf + (size_t)mat * kWELE;

  v8f acc[6];
  const v8f zero = {0.f, 0.f, 0.f, 0.f, 0.f, 0.f, 0.f, 0.f};
  #pragma unroll
  for (int j = 0; j < 6; ++j) acc[j] = zero;

  for (int kt = 0; kt < kD; kt += 32) {
    // A fragment (16x32 bf16): lanes 0-15 -> K = 0..7 / 16..23 (hi=0),
    // lanes 16-31 -> K = 8..15 / 24..31 (hi=1). Two contiguous b128 LDS loads.
    union { v16bf v; int4 q[2]; } A;
    const unsigned short* ap = &Xs[nl * kXS_STRIDE + kt + hi * 8];
    A.q[0] = *(const int4*)(ap);        // K = kt + hi*8 + 0..7      -> elems 0..7
    A.q[1] = *(const int4*)(ap + 16);   // K = kt + 16 + hi*8 + 0..7 -> elems 8..15

    #pragma unroll
    for (int j = 0; j < 6; ++j) {
      // B fragment (32x16 bf16): lane n = nl; K = hi*8 + 0..7 / 16 + hi*8 + 0..7 (matches A).
      int ncol = (wave * 6 + j) * 16 + nl;
      const unsigned short* bp = Wm + (size_t)ncol * kD + kt + hi * 8;
      union { v16bf v; int4 q[2]; } Bv;
      Bv.q[0] = *(const int4*)(bp);
      Bv.q[1] = *(const int4*)(bp + 16);
      acc[j] = __builtin_amdgcn_wmma_f32_16x16x32_bf16(
          false, A.v, false, Bv.v, (short)0, acc[j], false, false);
    }
  }

  // Epilogue: add bias, store fp32. C layout: VGPR r, lane L -> row r+8*(L>>4), col L&15.
  const float* bias = (mat == 0) ? bq : ((mat == 1) ? bk : bv);
  float* outm = qkv + (size_t)mat * kSZ;
  #pragma unroll
  for (int j = 0; j < 6; ++j) {
    int ncol = (wave * 6 + j) * 16 + nl;
    float bb = bias[ncol];
    size_t rbase = (size_t)mtile * 16 + hi * 8;
    #pragma unroll
    for (int r = 0; r < 8; ++r) {
      outm[(rbase + r) * kD + ncol] = acc[j][r] + bb;
    }
  }
}

// ---------------- Kernel 2: grid construction + bilinear sample + gated attention ----------------
// One block per (b, pout). 256 threads cover D=768 with 3 elems/thread.
__global__ __launch_bounds__(256) void k_attn(const float* __restrict__ qkv,
                                              const int* __restrict__ img_ids,
                                              const float* __restrict__ norm_x,
                                              const float* __restrict__ norm_y,
                                              const float* __restrict__ avgs,
                                              const float* __restrict__ stds,
                                              float* __restrict__ out) {
  __shared__ float red[256];
  const int bp   = blockIdx.x;
  const int b    = bp / kP;
  const int pout = bp - b * kP;
  const int t    = threadIdx.x;
  const int img  = img_ids[b];

  // grid value at flat channel index f in [0, 2P): reproduces the (B,2,P)->(B,gd,gd,2) reshape.
  auto gval = [&](int f) -> float {
    int ch = (f >= kP) ? 1 : 0;
    int pp = f - ch * kP;
    const float* nb = ch ? norm_y : norm_x;
    float a = nb[b * kP + pp] + avgs[((size_t)img * 2 + ch) * kP + pp];
    float s = stds[((size_t)img * 2 + ch) * kP + pp];
    return tanhf(a * s);
  };
  float gx = gval(2 * pout);
  float gy = gval(2 * pout + 1);
  float ix = ((gx + 1.f) * (float)kGD - 1.f) * 0.5f;
  float iy = ((gy + 1.f) * (float)kGD - 1.f) * 0.5f;
  float x0f = floorf(ix), y0f = floorf(iy);
  float x1f = x0f + 1.f,  y1f = y0f + 1.f;
  float wx1 = ix - x0f, wx0 = 1.f - wx1;
  float wy1 = iy - y0f, wy0 = 1.f - wy1;

  auto corner = [&](float xf, float yf, float& w) -> int {
    bool valid = (xf >= 0.f) && (xf <= (float)(kGD - 1)) &&
                 (yf >= 0.f) && (yf <= (float)(kGD - 1));
    int xi = (int)fminf(fmaxf(xf, 0.f), (float)(kGD - 1));
    int yi = (int)fminf(fmaxf(yf, 0.f), (float)(kGD - 1));
    if (!valid) w = 0.f;
    return yi * kGD + xi;
  };
  float w00 = wx0 * wy0, w10 = wx1 * wy0, w01 = wx0 * wy1, w11 = wx1 * wy1;
  int c00 = corner(x0f, y0f, w00);
  int c10 = corner(x1f, y0f, w10);
  int c01 = corner(x0f, y1f, w01);
  int c11 = corner(x1f, y1f, w11);

  const float* Q = qkv;
  const float* K = qkv + kSZ;
  const float* V = qkv + 2 * kSZ;
  const float* Qr  = Q + ((size_t)b * kP + pout) * kD;
  const float* Kb0 = K + (size_t)b * kP * kD;
  const float* Vb0 = V + (size_t)b * kP * kD;
  const float* K00 = Kb0 + (size_t)c00 * kD;
  const float* K10 = Kb0 + (size_t)c10 * kD;
  const float* K01 = Kb0 + (size_t)c01 * kD;
  const float* K11 = Kb0 + (size_t)c11 * kD;
  const float* V00 = Vb0 + (size_t)c00 * kD;
  const float* V10 = Vb0 + (size_t)c10 * kD;
  const float* V01 = Vb0 + (size_t)c01 * kD;
  const float* V11 = Vb0 + (size_t)c11 * kD;

  float partial = 0.f;
  float sv[3];
  #pragma unroll
  for (int i = 0; i < 3; ++i) {
    int d = t + i * 256;
    float sk   = w00 * K00[d] + w10 * K10[d] + w01 * K01[d] + w11 * K11[d];
    float sval = w00 * V00[d] + w10 * V10[d] + w01 * V01[d] + w11 * V11[d];
    partial += sk * Qr[d];
    sv[i] = sval;
  }
  red[t] = partial;
  __syncthreads();
  for (int sN = 128; sN > 0; sN >>= 1) {
    if (t < sN) red[t] += red[t + sN];
    __syncthreads();
  }
  float sig = 1.f / (1.f + __expf(-0.01f * red[0]));
  float* orow = out + ((size_t)b * kP + pout) * kD;
  #pragma unroll
  for (int i = 0; i < 3; ++i) orow[t + i * 256] = sig * sv[i];
}

extern "C" void kernel_launch(void* const* d_in, const int* in_sizes, int n_in,
                              void* d_out, int out_size, void* d_ws, size_t ws_size,
                              hipStream_t stream) {
  const float* x       = (const float*)d_in[0];
  const int*   img_ids = (const int*)  d_in[1];
  // d_in[2] = mask (unused by the reference)
  const float* norm_x  = (const float*)d_in[3];
  const float* norm_y  = (const float*)d_in[4];
  const float* Wq      = (const float*)d_in[5];
  const float* bq      = (const float*)d_in[6];
  const float* Wk      = (const float*)d_in[7];
  const float* bk      = (const float*)d_in[8];
  const float* Wv      = (const float*)d_in[9];
  const float* bv      = (const float*)d_in[10];
  const float* avgs    = (const float*)d_in[11];
  const float* stds    = (const float*)d_in[12];

  unsigned short* Wbf = (unsigned short*)d_ws;
  float* qkv = (float*)((char*)d_ws + kQKV_BYTE_OFF);
  float* outp = (float*)d_out;

  // 1) W -> bf16 (3*589824 elements, exactly 6912 blocks of 256)
  k_convw<<<(3 * kWELE) / 256, 256, 0, stream>>>(Wq, Wk, Wv, Wbf);

  // 2) fused QKV GEMM (bf16 WMMA, fp32 accumulate)
  dim3 g1(kM / 16, 3);
  k_gemm_qkv<<<g1, 256, 0, stream>>>(x, Wbf, bq, bk, bv, qkv);

  // 3) grid-sample + sigmoid-gated value
  k_attn<<<kM, 256, 0, stream>>>(qkv, img_ids, norm_x, norm_y, avgs, stds, outp);
}
